// diff_CSDI_22093311770891
// MI455X (gfx1250) — matrix operand order repaired and checked by reading, not compiled, not run on hardware
//
#include <hip/hip_runtime.h>
#include <hip/hip_bf16.h>
#include <math.h>

// ---------------------------------------------------------------------------
// CSDI diffusion forward for MI455X (gfx1250, wave32, WMMA).
// All dense projections run through v_wmma_f32_16x16x32_f16 (f32->f16 staged
// in LDS, f32 accumulate). Attention (head dims 2/8) is fused online-softmax
// on VALU since those shapes cannot fill a 16x16x32 matrix tile.
//
// Input order (setup_inputs dict): x, cond_info, diffusion_step, cfg_mask,
// timestep_emb, context, then params flattened as a pytree with dict keys in
// sorted order (jax tree_flatten convention): 234 leaves.
// ---------------------------------------------------------------------------

#define Bn    4
#define Kn    32
#define Ln    192
#define Cn    64
#define DTn   16
#define Hn    8
#define EMBn  128
#define CTXLn 64
#define CTXDn 768
#define SIDEn 145
#define KLn   (Kn * Ln)        // 6144
#define TOK   (Bn * Kn * Ln)   // 24576

typedef _Float16 v16h __attribute__((ext_vector_type(16)));
typedef _Float16 h8   __attribute__((ext_vector_type(8)));
typedef float    v8f  __attribute__((ext_vector_type(8)));

// ----------------------------- device helpers ------------------------------

__device__ __forceinline__ float gelu_f(float x) {
    // tanh approximation (jax.nn.gelu default approximate=True)
    float x3 = x * x * x;
    return 0.5f * x * (1.0f + tanhf(0.7978845608028654f * (x + 0.044715f * x3)));
}
__device__ __forceinline__ float sigmoid_f(float x) {
    return 1.0f / (1.0f + __expf(-x));
}

template <int ACT>
__device__ __forceinline__ float act_f(float x) {
    if (ACT == 1) return fmaxf(x, 0.0f);   // relu
    if (ACT == 2) return gelu_f(x);        // gelu
    return x;
}

__device__ __forceinline__ h8 cvt8(float4 f0, float4 f1, float s) {
    h8 hv;
    hv[0] = (_Float16)(f0.x * s); hv[1] = (_Float16)(f0.y * s);
    hv[2] = (_Float16)(f0.z * s); hv[3] = (_Float16)(f0.w * s);
    hv[4] = (_Float16)(f1.x * s); hv[5] = (_Float16)(f1.y * s);
    hv[6] = (_Float16)(f1.z * s); hv[7] = (_Float16)(f1.w * s);
    return hv;
}

// ------------------------------ WMMA GEMM ----------------------------------
// out[m,n] = act( sum_k A[m,k]*ascale * W[k,n or n,k] + bias[n] + add[m,n] )
// Block: 256 threads = 8 wave32 waves; 64x64 output tile; 32-wide K steps.
// Each wave owns a 16(row) x 32(col) sub-tile => two 16x16 WMMA accumulators.
// KV (compile-time) = K%8==0: staging is pure vector-load-or-zero (no scalar
// bounds ladder in the binary). KV=false only for the ragged K=145 cond conv.

template <bool WT, int ACT, bool KV>
__global__ __launch_bounds__(256)
void gemm_wmma(const float* __restrict__ A, const float* __restrict__ W,
               const float* __restrict__ bias, const float* __restrict__ add,
               float* __restrict__ outp, int M, int N, int K, float ascale) {
    __shared__ __align__(16) _Float16 sA[64][32];   // [m][k]
    __shared__ __align__(16) _Float16 sB[64][32];   // [n][k] (transposed W tile)

    const int tid  = threadIdx.x;
    const int wave = tid >> 5;
    const int lane = tid & 31;
    const int wrow = wave >> 1;       // 0..3  (16-row slice)
    const int wcol = wave & 1;        // 0..1  (32-col slice)
    const int bm   = blockIdx.x * 64;
    const int bn   = blockIdx.y * 64;
    const int half = lane >> 4;       // 0/1
    const int lrow = lane & 15;

    // per-thread staging coordinates
    const int a_row  = tid >> 2;             // 0..63
    const int a_koff = (tid & 3) * 8;        // 0,8,16,24
    const int b_k    = tid >> 3;             // 0..31  (!WT path)
    const int b_noff = (tid & 7) * 8;        // 0..56  (!WT path)

    v8f acc0 = {};
    v8f acc1 = {};

    for (int k0 = 0; k0 < K; k0 += 32) {
        // ---- stage A tile (f32 -> f16), 8 contiguous k per thread ----
        {
            const int gm = bm + a_row;
            const int gk = k0 + a_koff;
            h8 hv = {};
            if (KV) {
                // K%8==0 and gk%8==0: gk<K implies whole chunk in-bounds
                if (gm < M && gk < K) {
                    const float* ap = A + (size_t)gm * K + gk;
                    float4 f0 = *(const float4*)(ap);
                    float4 f1 = *(const float4*)(ap + 4);
                    hv = cvt8(f0, f1, ascale);
                    if (k0 + 32 < K) __builtin_prefetch(ap + 32, 0, 0);
                }
            } else {
#pragma unroll
                for (int j = 0; j < 8; j++) {
                    float v = (gm < M && (gk + j) < K)
                                  ? A[(size_t)gm * K + gk + j] * ascale : 0.0f;
                    hv[j] = (_Float16)v;
                }
            }
            *(h8*)&sA[a_row][a_koff] = hv;
        }
        // ---- stage W tile transposed to [n][k] ----
        if (WT) {
            // W[n,k]: contiguous along k, same scheme as A
            const int gn = bn + a_row;
            const int gk = k0 + a_koff;
            h8 hv = {};
            if (KV) {
                if (gn < N && gk < K) {
                    const float* wp = W + (size_t)gn * K + gk;
                    float4 f0 = *(const float4*)(wp);
                    float4 f1 = *(const float4*)(wp + 4);
                    hv = cvt8(f0, f1, 1.0f);
                }
            } else {
#pragma unroll
                for (int j = 0; j < 8; j++) {
                    float v = (gn < N && (gk + j) < K)
                                  ? W[(size_t)gn * K + gk + j] : 0.0f;
                    hv[j] = (_Float16)v;
                }
            }
            *(h8*)&sB[a_row][a_koff] = hv;
        } else {
            // W[k,n]: contiguous along n (N is always a multiple of 8 here);
            // vector-load 8 n or zero-fill, scatter b16 into [n][k]
            const int gk = k0 + b_k;
            const int gn = bn + b_noff;
            h8 hv = {};
            if (gk < K && gn < N) {
                const float* wp = W + (size_t)gk * N + gn;
                float4 f0 = *(const float4*)(wp);
                float4 f1 = *(const float4*)(wp + 4);
                hv = cvt8(f0, f1, 1.0f);
            }
#pragma unroll
            for (int j = 0; j < 8; j++) sB[b_noff + j][b_k] = hv[j];
        }
        __syncthreads();

        // A fragment: lane (m = wrow*16+lrow). elems 0..7 -> k = half*8+e,
        // elems 8..15 -> k = 16 + half*8 + (e-8). Two contiguous 8-half loads.
        v16h af, b0f, b1f;
        {
            const _Float16* ar = &sA[wrow * 16 + lrow][0];
            h8 lo = *(const h8*)(ar + half * 8);
            h8 hi = *(const h8*)(ar + 16 + half * 8);
#pragma unroll
            for (int e = 0; e < 8; e++) { af[e] = lo[e]; af[8 + e] = hi[e]; }
        }
        // B fragment: lane (n = base+lrow). same k order as A fragment.
        {
            const _Float16* br = &sB[wcol * 32 + lrow][0];
            h8 lo = *(const h8*)(br + half * 8);
            h8 hi = *(const h8*)(br + 16 + half * 8);
#pragma unroll
            for (int e = 0; e < 8; e++) { b0f[e] = lo[e]; b0f[8 + e] = hi[e]; }
            const _Float16* br2 = &sB[wcol * 32 + 16 + lrow][0];
            h8 lo2 = *(const h8*)(br2 + half * 8);
            h8 hi2 = *(const h8*)(br2 + 16 + half * 8);
#pragma unroll
            for (int e = 0; e < 8; e++) { b1f[e] = lo2[e]; b1f[8 + e] = hi2[e]; }
        }
        acc0 = __builtin_amdgcn_wmma_f32_16x16x32_f16(false, af, false, b0f,
                                                      (short)0, acc0, false, false);
        acc1 = __builtin_amdgcn_wmma_f32_16x16x32_f16(false, af, false, b1f,
                                                      (short)0, acc1, false, false);
        __syncthreads();
    }

    // epilogue: C/D layout — vgpr v: m = v (lanes 0-15) / v+8 (lanes 16-31),
    // n = lane&15.
#pragma unroll
    for (int nt = 0; nt < 2; nt++) {
        int gn = bn + wcol * 32 + nt * 16 + lrow;
        if (gn >= N) continue;
        float bv = bias ? bias[gn] : 0.0f;
#pragma unroll
        for (int v = 0; v < 8; v++) {
            int gm = bm + wrow * 16 + v + half * 8;
            if (gm >= M) continue;
            float r = (nt ? acc1[v] : acc0[v]) + bv;
            if (add) r += add[(size_t)gm * N + gn];
            outp[(size_t)gm * N + gn] = act_f<ACT>(r);
        }
    }
}

// ------------------------- elementwise / small kernels ----------------------

__global__ void k_zero(float* p, int n) {
    int i = blockIdx.x * 256 + threadIdx.x;
    if (i < n) p[i] = 0.0f;
}

// input conv (Cin=2) + relu, direct to token-major (t, 64)
__global__ void k_in_conv(const float* __restrict__ x, const float* __restrict__ w,
                          const float* __restrict__ b, float* __restrict__ xc) {
    int idx = blockIdx.x * 256 + threadIdx.x;
    if (idx >= TOK * 64) return;
    int t = idx >> 6, c = idx & 63;
    int bb = t / KLn, kl = t % KLn;
    float v0 = x[(size_t)(bb * 2 + 0) * KLn + kl];
    float v1 = x[(size_t)(bb * 2 + 1) * KLn + kl];
    float r = w[c * 2 + 0] * v0 + w[c * 2 + 1] * v1 + b[c];
    xc[idx] = fmaxf(r, 0.0f);
}

// diffusion embedding MLP: table row (sin/cos) -> silu(@w1) -> silu(@w2)
__global__ void k_diff_emb(const int* __restrict__ step,
                           const float* __restrict__ w1, const float* __restrict__ b1,
                           const float* __restrict__ w2, const float* __restrict__ b2,
                           float* __restrict__ e2out) {
    int bb = blockIdx.x;
    int j  = threadIdx.x;   // 0..127
    __shared__ float e0[EMBn], e1[EMBn];
    float s = (float)step[bb];
    {
        int jj = j & 63;
        float f = __powf(10.0f, (float)jj / 63.0f * 4.0f);
        float t = s * f;
        e0[j] = (j < 64) ? __sinf(t) : __cosf(t);
    }
    __syncthreads();
    float a = b1[j];
    for (int e = 0; e < EMBn; e++) a += e0[e] * w1[e * EMBn + j];
    a = a * sigmoid_f(a);
    e1[j] = a;
    __syncthreads();
    float a2 = b2[j];
    for (int e = 0; e < EMBn; e++) a2 += e1[e] * w2[e * EMBn + j];
    a2 = a2 * sigmoid_f(a2);
    e2out[bb * EMBn + j] = a2;
}

// dpe[b,c] = e2[b,:] @ dp_w[c,:] + dp_b[c]
__global__ void k_dp(const float* __restrict__ e2, const float* __restrict__ w,
                     const float* __restrict__ b, float* __restrict__ dpe) {
    int bb = blockIdx.x, c = threadIdx.x;
    float acc = b[c];
    for (int e = 0; e < EMBn; e++) acc += e2[bb * EMBn + e] * w[c * EMBn + e];
    dpe[bb * Cn + c] = acc;
}

__global__ void k_add_bcast(const float* __restrict__ xc, const float* __restrict__ dpe,
                            float* __restrict__ y) {
    int idx = blockIdx.x * 256 + threadIdx.x;
    if (idx >= TOK * 64) return;
    int t = idx >> 6, c = idx & 63;
    int bb = t / KLn;
    y[idx] = xc[idx] + dpe[bb * Cn + c];
}

// timestep_emb (B,DT,K,L) -> token-major (t,16)
__global__ void k_tte_perm(const float* __restrict__ te, float* __restrict__ tte) {
    int idx = blockIdx.x * 256 + threadIdx.x;
    if (idx >= TOK * DTn) return;
    int t = idx / DTn, d = idx % DTn;
    int bb = t / KLn, kl = t % KLn;
    tte[idx] = te[(size_t)(bb * DTn + d) * KLn + kl];
}

// cond_info (B,SIDE,K,L) -> token-major (t,145)
__global__ void k_cond_perm(const float* __restrict__ cond, float* __restrict__ ct) {
    int idx = blockIdx.x * 256 + threadIdx.x;
    if (idx >= TOK * SIDEn) return;
    int t = idx / SIDEn, s = idx % SIDEn;
    int bb = t / KLn, kl = t % KLn;
    ct[idx] = cond[(size_t)(bb * SIDEn + s) * KLn + kl];
}

// context * cfg_mask[b]
__global__ void k_mask_ctx(const float* __restrict__ ctx, const float* __restrict__ mask,
                           float* __restrict__ out) {
    int idx = blockIdx.x * 256 + threadIdx.x;
    if (idx >= Bn * CTXDn * CTXLn) return;
    int bb = idx / (CTXDn * CTXLn);
    out[idx] = ctx[idx] * mask[bb];
}

// chan conv: c_tok[(b*64+j), o] = sum_c chan_w[o,c]*ctxm[b,c,j] + chan_b[o]
__global__ void k_ctok(const float* __restrict__ ctxm, const float* __restrict__ w,
                       const float* __restrict__ b, float* __restrict__ c_tok) {
    int bb = blockIdx.x, j = blockIdx.y, o = threadIdx.x;
    const float* base = ctxm + (size_t)bb * CTXDn * CTXLn + j;
    float acc = b[o];
    for (int c = 0; c < CTXDn; c++) acc += w[o * CTXDn + c] * base[(size_t)c * CTXLn];
    c_tok[((size_t)bb * CTXLn + j) * Cn + o] = acc;
}

// LN(a + r) over last dim D, token per thread
template <int D>
__global__ void k_add_ln(const float* __restrict__ a, const float* __restrict__ r,
                         const float* __restrict__ gamma, const float* __restrict__ beta,
                         float* __restrict__ out, int ntok) {
    int t = blockIdx.x * 256 + threadIdx.x;
    if (t >= ntok) return;
    float v[D];
    float mean = 0.0f;
#pragma unroll
    for (int i = 0; i < D; i++) {
        v[i] = a[(size_t)t * D + i] + r[(size_t)t * D + i];
        mean += v[i];
    }
    mean *= (1.0f / D);
    float var = 0.0f;
#pragma unroll
    for (int i = 0; i < D; i++) {
        float d = v[i] - mean;
        var += d * d;
    }
    var *= (1.0f / D);
    float inv = rsqrtf(var + 1e-5f);
#pragma unroll
    for (int i = 0; i < D; i++)
        out[(size_t)t * D + i] = (v[i] - mean) * inv * gamma[i] + beta[i];
}

// TV attention: grid (B*K, H), block 192 (query per thread). head dim 2.
__global__ void k_tv_attn(const float* __restrict__ qt, const float* __restrict__ kt,
                          const float* __restrict__ vt, const float* __restrict__ vv,
                          float* __restrict__ ot, float* __restrict__ ov) {
    int n = blockIdx.x, h = blockIdx.y;
    int i = threadIdx.x;  // query 0..191
    __shared__ float ks[Ln][2], vts[Ln][2], vvs[Ln][8];
    const size_t b16 = (size_t)n * Ln * DTn;
    const size_t b64 = (size_t)n * Ln * Cn;
    ks[i][0]  = kt[b16 + (size_t)i * DTn + h * 2 + 0];
    ks[i][1]  = kt[b16 + (size_t)i * DTn + h * 2 + 1];
    vts[i][0] = vt[b16 + (size_t)i * DTn + h * 2 + 0];
    vts[i][1] = vt[b16 + (size_t)i * DTn + h * 2 + 1];
#pragma unroll
    for (int d = 0; d < 8; d++)
        vvs[i][d] = vv[b64 + (size_t)i * Cn + h * 8 + d];
    __syncthreads();

    float q0 = qt[b16 + (size_t)i * DTn + h * 2 + 0];
    float q1 = qt[b16 + (size_t)i * DTn + h * 2 + 1];
    const float scale = 0.70710678118f;  // 1/sqrt(2)
    float m = -1e30f, s = 0.0f;
    float at0 = 0.0f, at1 = 0.0f;
    float av[8] = {0, 0, 0, 0, 0, 0, 0, 0};
    for (int j = 0; j < Ln; j++) {
        float sc = (q0 * ks[j][0] + q1 * ks[j][1]) * scale;
        float mn = fmaxf(m, sc);
        float corr = __expf(m - mn);
        float p = __expf(sc - mn);
        s = s * corr + p;
        at0 = at0 * corr + p * vts[j][0];
        at1 = at1 * corr + p * vts[j][1];
#pragma unroll
        for (int d = 0; d < 8; d++) av[d] = av[d] * corr + p * vvs[j][d];
        m = mn;
    }
    float inv = 1.0f / s;
    ot[b16 + (size_t)i * DTn + h * 2 + 0] = at0 * inv;
    ot[b16 + (size_t)i * DTn + h * 2 + 1] = at1 * inv;
#pragma unroll
    for (int d = 0; d < 8; d++)
        ov[b64 + (size_t)i * Cn + h * 8 + d] = av[d] * inv;
}

// feature attention: grid (B*L, H), block 32 (query per thread). head dim 8.
__global__ void k_feat_attn(const float* __restrict__ q, const float* __restrict__ k,
                            const float* __restrict__ v, float* __restrict__ o) {
    int n = blockIdx.x, h = blockIdx.y;
    int i = threadIdx.x;  // query 0..31
    __shared__ float ks[Kn][8], vs[Kn][8];
    const size_t base = (size_t)n * Kn * Cn;
#pragma unroll
    for (int d = 0; d < 8; d++) {
        ks[i][d] = k[base + (size_t)i * Cn + h * 8 + d];
        vs[i][d] = v[base + (size_t)i * Cn + h * 8 + d];
    }
    __syncthreads();
    float qv[8];
#pragma unroll
    for (int d = 0; d < 8; d++) qv[d] = q[base + (size_t)i * Cn + h * 8 + d];
    const float scale = 0.35355339059f;  // 1/sqrt(8)
    float m = -1e30f, s = 0.0f, acc[8] = {0, 0, 0, 0, 0, 0, 0, 0};
    for (int j = 0; j < Kn; j++) {
        float sc = 0.0f;
#pragma unroll
        for (int d = 0; d < 8; d++) sc += qv[d] * ks[j][d];
        sc *= scale;
        float mn = fmaxf(m, sc);
        float corr = __expf(m - mn);
        float p = __expf(sc - mn);
        s = s * corr + p;
#pragma unroll
        for (int d = 0; d < 8; d++) acc[d] = acc[d] * corr + p * vs[j][d];
        m = mn;
    }
    float inv = 1.0f / s;
#pragma unroll
    for (int d = 0; d < 8; d++)
        o[base + (size_t)i * Cn + h * 8 + d] = acc[d] * inv;
}

// cross attention: grid (B, H, 6144/256), block 256. 64 kv tokens, head dim 8.
__global__ void k_cross_attn(const float* __restrict__ q, const float* __restrict__ kc,
                             const float* __restrict__ vc, float* __restrict__ o) {
    int bb = blockIdx.x, h = blockIdx.y;
    int qi = blockIdx.z * 256 + threadIdx.x;  // 0..6143
    __shared__ float ks[CTXLn][8], vs[CTXLn][8];
    for (int idx = threadIdx.x; idx < CTXLn * 8; idx += 256) {
        int j = idx >> 3, d = idx & 7;
        ks[j][d] = kc[((size_t)bb * CTXLn + j) * Cn + h * 8 + d];
        vs[j][d] = vc[((size_t)bb * CTXLn + j) * Cn + h * 8 + d];
    }
    __syncthreads();
    const size_t qb = ((size_t)bb * KLn + qi) * Cn + h * 8;
    float qv[8];
#pragma unroll
    for (int d = 0; d < 8; d++) qv[d] = q[qb + d];
    const float scale = 0.35355339059f;  // 1/sqrt(8)
    float m = -1e30f, s = 0.0f, acc[8] = {0, 0, 0, 0, 0, 0, 0, 0};
    for (int j = 0; j < CTXLn; j++) {
        float sc = 0.0f;
#pragma unroll
        for (int d = 0; d < 8; d++) sc += qv[d] * ks[j][d];
        sc *= scale;
        float mn = fmaxf(m, sc);
        float corr = __expf(m - mn);
        float p = __expf(sc - mn);
        s = s * corr + p;
#pragma unroll
        for (int d = 0; d < 8; d++) acc[d] = acc[d] * corr + p * vs[j][d];
        m = mn;
    }
    float inv = 1.0f / s;
#pragma unroll
    for (int d = 0; d < 8; d++) o[qb + d] = acc[d] * inv;
}

// token permutes: TV order t=((b*K+k)*L+l)  <->  feature order t'=((b*L+l)*K+k)
__global__ void k_perm_tv2feat(const float* __restrict__ in, float* __restrict__ out) {
    int idx = blockIdx.x * 256 + threadIdx.x;
    if (idx >= TOK * 64) return;
    int tp = idx >> 6, c = idx & 63;  // dst (feat order)
    int bb = tp / (Ln * Kn);
    int r = tp % (Ln * Kn);
    int l = r / Kn, k = r % Kn;
    int t = (bb * Kn + k) * Ln + l;
    out[idx] = in[(size_t)t * 64 + c];
}
__global__ void k_perm_feat2tv(const float* __restrict__ in, float* __restrict__ out) {
    int idx = blockIdx.x * 256 + threadIdx.x;
    if (idx >= TOK * 64) return;
    int t = idx >> 6, c = idx & 63;   // dst (tv order)
    int bb = t / (Kn * Ln);
    int r = t % (Kn * Ln);
    int k = r / Ln, l = r % Ln;
    int tp = (bb * Ln + l) * Kn + k;
    out[idx] = in[(size_t)tp * 64 + c];
}

// gated activation: sigmoid(gate)*tanh(filt), in (t,128) -> (t,64)
__global__ void k_gate(const float* __restrict__ y128, float* __restrict__ out) {
    int idx = blockIdx.x * 256 + threadIdx.x;
    if (idx >= TOK * 64) return;
    int t = idx >> 6, c = idx & 63;
    float g = y128[(size_t)t * 128 + c];
    float f = y128[(size_t)t * 128 + 64 + c];
    out[idx] = sigmoid_f(g) * tanhf(f);
}

// residual/skip: xc=(xc+res)/sqrt2 ; skip+=skippart
__global__ void k_res_skip(float* __restrict__ xc, float* __restrict__ skip,
                           const float* __restrict__ y128) {
    int idx = blockIdx.x * 256 + threadIdx.x;
    if (idx >= TOK * 64) return;
    int t = idx >> 6, c = idx & 63;
    float res = y128[(size_t)t * 128 + c];
    float sk  = y128[(size_t)t * 128 + 64 + c];
    xc[idx]   = (xc[idx] + res) * 0.70710678118f;
    skip[idx] = skip[idx] + sk;
}

// final 64->1 projection
__global__ void k_final(const float* __restrict__ s1, const float* __restrict__ w,
                        const float* __restrict__ b, float* __restrict__ out) {
    int t = blockIdx.x * 256 + threadIdx.x;
    if (t >= TOK) return;
    float acc = b[0];
#pragma unroll
    for (int c = 0; c < 64; c++) acc += s1[(size_t)t * 64 + c] * w[c];
    out[t] = acc;
}

// ------------------------------- host side ---------------------------------

struct LayerP {
    const float *chan_b, *chan_w, *cond_b, *cond_w;
    const float *cr_ffn_b1, *cr_ffn_b2, *cr_ffn_w1, *cr_ffn_w2;
    const float *cr_ln1_b, *cr_ln1_s, *cr_ln2_b, *cr_ln2_s;
    const float *cr_wk, *cr_wo, *cr_wq, *cr_wv;
    const float *dp_b, *dp_w;
    const float *ft_ffn_b1, *ft_ffn_b2, *ft_ffn_w1, *ft_ffn_w2;
    const float *ft_ln1_b, *ft_ln1_s, *ft_ln2_b, *ft_ln2_s;
    const float *ft_wk, *ft_wo, *ft_wq, *ft_wv;
    const float *mid_b, *mid_w, *out_b, *out_w;
    const float *t_ffnt_b1, *t_ffnt_b2, *t_ffnt_w1, *t_ffnt_w2;
    const float *t_ffnv_b1, *t_ffnv_b2, *t_ffnv_w1, *t_ffnv_w2;
    const float *t_lnt1_b, *t_lnt1_s, *t_lnt2_b, *t_lnt2_s;
    const float *t_lnv1_b, *t_lnv1_s, *t_lnv2_b, *t_lnv2_s;
    const float *t_wk, *t_wot, *t_wov, *t_wq, *t_wvt, *t_wvv;
};
struct ParamsP {
    const float *de_b1, *de_b2, *de_w1, *de_w2, *in_b, *in_w;
    LayerP layer[4];
    const float *o1_b, *o1_w, *o2_b, *o2_w;
};

static void build_params(void* const* d_in, ParamsP& P) {
    int i = 6;
    auto nx = [&]() { return (const float*)d_in[i++]; };
    P.de_b1 = nx(); P.de_b2 = nx(); P.de_w1 = nx(); P.de_w2 = nx();
    P.in_b = nx();  P.in_w = nx();
    for (int l = 0; l < 4; l++) {
        LayerP& L = P.layer[l];
        L.chan_b = nx(); L.chan_w = nx(); L.cond_b = nx(); L.cond_w = nx();
        L.cr_ffn_b1 = nx(); L.cr_ffn_b2 = nx(); L.cr_ffn_w1 = nx(); L.cr_ffn_w2 = nx();
        L.cr_ln1_b = nx(); L.cr_ln1_s = nx(); L.cr_ln2_b = nx(); L.cr_ln2_s = nx();
        L.cr_wk = nx(); L.cr_wo = nx(); L.cr_wq = nx(); L.cr_wv = nx();
        L.dp_b = nx(); L.dp_w = nx();
        L.ft_ffn_b1 = nx(); L.ft_ffn_b2 = nx(); L.ft_ffn_w1 = nx(); L.ft_ffn_w2 = nx();
        L.ft_ln1_b = nx(); L.ft_ln1_s = nx(); L.ft_ln2_b = nx(); L.ft_ln2_s = nx();
        L.ft_wk = nx(); L.ft_wo = nx(); L.ft_wq = nx(); L.ft_wv = nx();
        L.mid_b = nx(); L.mid_w = nx(); L.out_b = nx(); L.out_w = nx();
        L.t_ffnt_b1 = nx(); L.t_ffnt_b2 = nx(); L.t_ffnt_w1 = nx(); L.t_ffnt_w2 = nx();
        L.t_ffnv_b1 = nx(); L.t_ffnv_b2 = nx(); L.t_ffnv_w1 = nx(); L.t_ffnv_w2 = nx();
        L.t_lnt1_b = nx(); L.t_lnt1_s = nx(); L.t_lnt2_b = nx(); L.t_lnt2_s = nx();
        L.t_lnv1_b = nx(); L.t_lnv1_s = nx(); L.t_lnv2_b = nx(); L.t_lnv2_s = nx();
        L.t_wk = nx(); L.t_wot = nx(); L.t_wov = nx();
        L.t_wq = nx(); L.t_wvt = nx(); L.t_wvv = nx();
    }
    P.o1_b = nx(); P.o1_w = nx(); P.o2_b = nx(); P.o2_w = nx();
}

extern "C" void kernel_launch(void* const* d_in, const int* in_sizes, int n_in,
                              void* d_out, int out_size, void* d_ws, size_t ws_size,
                              hipStream_t stream) {
    (void)in_sizes; (void)n_in; (void)out_size; (void)ws_size;

    const float* x_in   = (const float*)d_in[0];
    const float* cond_in= (const float*)d_in[1];
    const int*   dstep  = (const int*)d_in[2];
    const float* cmask  = (const float*)d_in[3];
    const float* temb   = (const float*)d_in[4];
    const float* ctx    = (const float*)d_in[5];
    float* out          = (float*)d_out;

    ParamsP P;
    build_params(d_in, P);

    // ---- workspace carve-up (floats) ----
    float* p = (float*)d_ws;
    auto alloc = [&](size_t n) { float* r = p; p += n; return r; };
    float* xc       = alloc((size_t)TOK * 64);
    float* skip     = alloc((size_t)TOK * 64);
    float* tte      = alloc((size_t)TOK * 16);
    float* e2       = alloc((size_t)Bn * EMBn);
    float* dpe      = alloc((size_t)Bn * Cn);
    float* cond_tok = alloc((size_t)TOK * SIDEn);
    float* ctxm     = alloc((size_t)Bn * CTXDn * CTXLn);
    float* c_tok    = alloc((size_t)Bn * CTXLn * Cn);
    float* kc       = alloc((size_t)Bn * CTXLn * Cn);
    float* vc       = alloc((size_t)Bn * CTXLn * Cn);
    float* q16      = alloc((size_t)TOK * 16);
    float* k16      = alloc((size_t)TOK * 16);
    float* v16t     = alloc((size_t)TOK * 16);
    float* o16a     = alloc((size_t)TOK * 16);
    float* o16b     = alloc((size_t)TOK * 16);
    float* bufA     = alloc((size_t)TOK * 64);
    float* bufB     = alloc((size_t)TOK * 64);
    float* bufC     = alloc((size_t)TOK * 64);
    float* bufD     = alloc((size_t)TOK * 64);
    float* bufE     = alloc((size_t)TOK * 64);
    float* b128a    = alloc((size_t)TOK * 128);
    float* b128b    = alloc((size_t)TOK * 128);

    const int EW64  = (TOK * 64 + 255) / 256;   // elementwise grids
    const int EW16  = (TOK * 16 + 255) / 256;
    const int EW145 = (TOK * SIDEn + 255) / 256;

    auto GEMM = [&](const float* A, const float* W, const float* bias,
                    const float* add, float* o, int M, int N, int K,
                    bool wt, int act, float ascale) {
        dim3 g((M + 63) / 64, (N + 63) / 64);
        const bool kv = ((K & 7) == 0);
        if (!wt && act == 0)
            gemm_wmma<false, 0, true><<<g, 256, 0, stream>>>(A, W, bias, add, o, M, N, K, ascale);
        else if (!wt && act == 2)
            gemm_wmma<false, 2, true><<<g, 256, 0, stream>>>(A, W, bias, add, o, M, N, K, ascale);
        else if (wt && act == 1)
            gemm_wmma<true, 1, true><<<g, 256, 0, stream>>>(A, W, bias, add, o, M, N, K, ascale);
        else if (wt && kv)
            gemm_wmma<true, 0, true><<<g, 256, 0, stream>>>(A, W, bias, add, o, M, N, K, ascale);
        else
            gemm_wmma<true, 0, false><<<g, 256, 0, stream>>>(A, W, bias, add, o, M, N, K, ascale);
    };

    // ---------- prologue ----------
    k_mask_ctx<<<(Bn * CTXDn * CTXLn + 255) / 256, 256, 0, stream>>>(ctx, cmask, ctxm);
    k_cond_perm<<<EW145, 256, 0, stream>>>(cond_in, cond_tok);
    k_tte_perm<<<EW16, 256, 0, stream>>>(temb, tte);
    k_in_conv<<<EW64, 256, 0, stream>>>(x_in, P.in_w, P.in_b, xc);
    k_zero<<<EW64, 256, 0, stream>>>(skip, TOK * 64);
    k_diff_emb<<<Bn, EMBn, 0, stream>>>(dstep, P.de_w1, P.de_b1, P.de_w2, P.de_b2, e2);

    // ---------- residual layers ----------
    for (int l = 0; l < 4; l++) {
        const LayerP& L = P.layer[l];

        k_dp<<<Bn, Cn, 0, stream>>>(e2, L.dp_w, L.dp_b, dpe);
        k_add_bcast<<<EW64, 256, 0, stream>>>(xc, dpe, bufA);   // y (yv)

        // ---- TV time attention (over L, batch B*K) ----
        GEMM(tte, L.t_wq, nullptr, nullptr, q16, TOK, 16, 16, false, 0, 1.f);
        GEMM(tte, L.t_wk, nullptr, nullptr, k16, TOK, 16, 16, false, 0, 1.f);
        GEMM(tte, L.t_wvt, nullptr, nullptr, v16t, TOK, 16, 16, false, 0, 1.f);
        GEMM(bufA, L.t_wvv, nullptr, nullptr, bufB, TOK, 64, 64, false, 0, 1.f);
        k_tv_attn<<<dim3(Bn * Kn, Hn), Ln, 0, stream>>>(q16, k16, v16t, bufB, o16a, bufC);
        GEMM(o16a, L.t_wot, nullptr, nullptr, o16b, TOK, 16, 16, false, 0, 1.f);
        GEMM(bufC, L.t_wov, nullptr, nullptr, bufD, TOK, 64, 64, false, 0, 1.f);
        k_add_ln<16><<<(TOK + 255) / 256, 256, 0, stream>>>(tte, o16b, L.t_lnt1_s, L.t_lnt1_b, q16, TOK);
        k_add_ln<64><<<(TOK + 255) / 256, 256, 0, stream>>>(bufA, bufD, L.t_lnv1_s, L.t_lnv1_b, bufB, TOK);
        // FFN (time stream): 16 -> 64 gelu -> 16
        GEMM(q16, L.t_ffnt_w1, L.t_ffnt_b1, nullptr, bufC, TOK, 64, 16, false, 2, 1.f);
        GEMM(bufC, L.t_ffnt_w2, L.t_ffnt_b2, nullptr, o16a, TOK, 16, 64, false, 0, 1.f);
        k_add_ln<16><<<(TOK + 255) / 256, 256, 0, stream>>>(q16, o16a, L.t_lnt2_s, L.t_lnt2_b, tte, TOK);
        // FFN (value stream): 64 -> 64 gelu -> 64
        GEMM(bufB, L.t_ffnv_w1, L.t_ffnv_b1, nullptr, bufC, TOK, 64, 64, false, 2, 1.f);
        GEMM(bufC, L.t_ffnv_w2, L.t_ffnv_b2, nullptr, bufD, TOK, 64, 64, false, 0, 1.f);
        k_add_ln<64><<<(TOK + 255) / 256, 256, 0, stream>>>(bufB, bufD, L.t_lnv2_s, L.t_lnv2_b, bufA, TOK);

        // ---- feature attention (over K, batch B*L) ----
        k_perm_tv2feat<<<EW64, 256, 0, stream>>>(bufA, bufB);   // yf
        GEMM(bufB, L.ft_wq, nullptr, nullptr, bufC, TOK, 64, 64, false, 0, 1.f);
        GEMM(bufB, L.ft_wk, nullptr, nullptr, bufD, TOK, 64, 64, false, 0, 1.f);
        GEMM(bufB, L.ft_wv, nullptr, nullptr, bufE, TOK, 64, 64, false, 0, 1.f);
        k_feat_attn<<<dim3(Bn * Ln, Hn), Kn, 0, stream>>>(bufC, bufD, bufE, bufA);
        GEMM(bufA, L.ft_wo, nullptr, nullptr, bufC, TOK, 64, 64, false, 0, 1.f);
        k_add_ln<64><<<(TOK + 255) / 256, 256, 0, stream>>>(bufB, bufC, L.ft_ln1_s, L.ft_ln1_b, bufD, TOK);
        GEMM(bufD, L.ft_ffn_w1, L.ft_ffn_b1, nullptr, bufC, TOK, 64, 64, false, 2, 1.f);
        GEMM(bufC, L.ft_ffn_w2, L.ft_ffn_b2, nullptr, bufE, TOK, 64, 64, false, 0, 1.f);
        k_add_ln<64><<<(TOK + 255) / 256, 256, 0, stream>>>(bufD, bufE, L.ft_ln2_s, L.ft_ln2_b, bufB, TOK);
        k_perm_feat2tv<<<EW64, 256, 0, stream>>>(bufB, bufA);   // y back in TV order

        // ---- cross attention to text context ----
        k_ctok<<<dim3(Bn, CTXLn), Cn, 0, stream>>>(ctxm, L.chan_w, L.chan_b, c_tok);
        GEMM(c_tok, L.cr_wk, nullptr, nullptr, kc, Bn * CTXLn, 64, 64, false, 0, 1.f);
        GEMM(c_tok, L.cr_wv, nullptr, nullptr, vc, Bn * CTXLn, 64, 64, false, 0, 1.f);
        GEMM(bufA, L.cr_wq, nullptr, nullptr, bufB, TOK, 64, 64, false, 0, 1.f);
        k_cross_attn<<<dim3(Bn, Hn, KLn / 256), 256, 0, stream>>>(bufB, kc, vc, bufC);
        GEMM(bufC, L.cr_wo, nullptr, nullptr, bufD, TOK, 64, 64, false, 0, 1.f);
        k_add_ln<64><<<(TOK + 255) / 256, 256, 0, stream>>>(bufA, bufD, L.cr_ln1_s, L.cr_ln1_b, bufB, TOK);
        GEMM(bufB, L.cr_ffn_w1, L.cr_ffn_b1, nullptr, bufC, TOK, 64, 64, false, 2, 1.f);
        GEMM(bufC, L.cr_ffn_w2, L.cr_ffn_b2, nullptr, bufD, TOK, 64, 64, false, 0, 1.f);
        k_add_ln<64><<<(TOK + 255) / 256, 256, 0, stream>>>(bufB, bufD, L.cr_ln2_s, L.cr_ln2_b, bufA, TOK);

        // ---- WaveNet-style gated residual ----
        GEMM(cond_tok, L.cond_w, L.cond_b, nullptr, b128a, TOK, 128, SIDEn, true, 0, 1.f);
        GEMM(bufA, L.mid_w, L.mid_b, b128a, b128b, TOK, 128, 64, true, 0, 1.f);
        k_gate<<<EW64, 256, 0, stream>>>(b128b, bufB);
        GEMM(bufB, L.out_w, L.out_b, nullptr, b128a, TOK, 128, 64, true, 0, 1.f);
        k_res_skip<<<EW64, 256, 0, stream>>>(xc, skip, b128a);
    }

    // ---------- epilogue: (skip/sqrt(4)) -> relu conv -> 64->1 ----------
    GEMM(skip, P.o1_w, P.o1_b, nullptr, bufA, TOK, 64, 64, true, 1, 0.5f);
    k_final<<<(TOK + 255) / 256, 256, 0, stream>>>(bufA, P.o2_w, P.o2_b, out);
}
